// DeepseekV2MLA_15728170238340
// MI455X (gfx1250) — matrix operand order repaired and checked by reading, not compiled, not run on hardware
//
#include <hip/hip_runtime.h>
#include <hip/hip_bf16.h>

// ---------------------------------------------------------------------------
// DeepseekV2 MLA forward for MI455X (gfx1250), wave32 + bf16 WMMA.
// B=1, S=2048, HID=2048, H=16, NOPE=128, ROPE=64, VD=128, QD=192,
// QLR=1536, KVLR=512.
//
// GEMM: 128x128 block tile, 8 wave32 in a 2(M)x4(N) grid, each wave owns a
// 64x32 accumulator = 8 x v_wmma_f32_16x16x32_bf16 per K-step. A/B tiles are
// double-buffered in LDS and filled with global_load_async_to_lds_b128
// (ASYNCcnt) when the source is bf16; f32 sources (attn/ctx) are converted
// on stage via float4 loads + packed ds stores.
// ---------------------------------------------------------------------------

typedef __attribute__((ext_vector_type(16))) unsigned short ushort16;
typedef __attribute__((ext_vector_type(8)))  unsigned short ushort8;
typedef __attribute__((ext_vector_type(16))) __bf16         v16bf;
typedef __attribute__((ext_vector_type(8)))  float          v8f;

#define S_LEN 2048
#define HEADS 16
#define NOPE_D 128
#define ROPE_D 64
#define QD_D 192
#define VD_D 128

__device__ __forceinline__ unsigned short f2bf(float f) {
    unsigned int u = __builtin_bit_cast(unsigned int, f);
    unsigned int r = 0x7FFFu + ((u >> 16) & 1u);   // round-to-nearest-even
    return (unsigned short)((u + r) >> 16);
}

// ---------------------------------------------------------------------------
// C[M,N] = A[M,K] * B[N,K]^T (+bias | causal-mask*scale)
//   AT = unsigned short (bf16 bits, async-copy path) or float (cvt-on-stage)
//   B is always bf16 bits (pre-converted weights / packed activations).
// grid = (ceil(N/128), M/128, batch). M, K multiples of 32; N any (clamped).
// ---------------------------------------------------------------------------
template <typename AT, bool HAS_BIAS, int EPI>
__global__ __launch_bounds__(256) void wmma_gemm(
    const AT* __restrict__ A, int lda, long long strideA,
    const unsigned short* __restrict__ B, int ldb, long long strideB,
    const float* __restrict__ bias,
    float* __restrict__ C, int ldc, long long strideC,
    int N, int K, float scale)
{
    __shared__ unsigned short ldsA[2][128 * 32];   // 8 KB x2
    __shared__ unsigned short ldsB[2][128 * 32];   // 8 KB x2

    A += (long long)blockIdx.z * strideA;
    B += (long long)blockIdx.z * strideB;
    C += (long long)blockIdx.z * strideC;

    const int tid  = threadIdx.x;
    const int lane = tid & 31;
    const int wave = tid >> 5;
    const int m0   = blockIdx.y * 128;
    const int n0   = blockIdx.x * 128;

    const int wm0 = (wave & 1) * 64;    // wave M offset within block tile
    const int wn0 = (wave >> 1) * 32;   // wave N offset within block tile

    // ---- stage B tile (128 rows x 32 bf16) via async copy: 2 x b128/thread
    auto stageB = [&](int buf, int k0) {
#pragma unroll
        for (int tdx = 0; tdx < 2; ++tdx) {
            int idx = tid * 2 + tdx;            // 0..511
            int row = idx >> 2;                 // 0..127
            int seg = idx & 3;                  // 16B segment in row
            int gn  = n0 + row;
            if (gn > N - 1) gn = N - 1;         // clamp (dup loads, safe)
            const unsigned short* gp = B + (size_t)gn * ldb + k0 + seg * 8;
            unsigned lo = (unsigned)(size_t)(&ldsB[buf][row * 32 + seg * 8]);
            unsigned long long ga = (unsigned long long)(size_t)gp;
            asm volatile("global_load_async_to_lds_b128 %0, %1, off"
                         :: "v"(lo), "v"(ga) : "memory");
        }
    };

    // ---- stage A tile: async when bf16, cvt-on-stage when f32
    auto stageA = [&](int buf, int k0) {
        if constexpr (sizeof(AT) == 2) {
#pragma unroll
            for (int tdx = 0; tdx < 2; ++tdx) {
                int idx = tid * 2 + tdx;
                int row = idx >> 2;
                int seg = idx & 3;
                const unsigned short* gp =
                    (const unsigned short*)A + (size_t)(m0 + row) * lda + k0 + seg * 8;
                unsigned lo = (unsigned)(size_t)(&ldsA[buf][row * 32 + seg * 8]);
                unsigned long long ga = (unsigned long long)(size_t)gp;
                asm volatile("global_load_async_to_lds_b128 %0, %1, off"
                             :: "v"(lo), "v"(ga) : "memory");
            }
        } else {
            int row  = tid >> 1;                // 0..127
            int colh = (tid & 1) * 16;          // 0 or 16
            const float4* fp = (const float4*)((const float*)A +
                               (size_t)(m0 + row) * lda + k0 + colh);
            float4 f0 = fp[0], f1 = fp[1], f2 = fp[2], f3 = fp[3];
            ushort8 s0, s1;
            s0[0] = f2bf(f0.x); s0[1] = f2bf(f0.y); s0[2] = f2bf(f0.z); s0[3] = f2bf(f0.w);
            s0[4] = f2bf(f1.x); s0[5] = f2bf(f1.y); s0[6] = f2bf(f1.z); s0[7] = f2bf(f1.w);
            s1[0] = f2bf(f2.x); s1[1] = f2bf(f2.y); s1[2] = f2bf(f2.z); s1[3] = f2bf(f2.w);
            s1[4] = f2bf(f3.x); s1[5] = f2bf(f3.y); s1[6] = f2bf(f3.z); s1[7] = f2bf(f3.w);
            *(ushort8*)(&ldsA[buf][row * 32 + colh])     = s0;
            *(ushort8*)(&ldsA[buf][row * 32 + colh + 8]) = s1;
        }
    };

    v8f zero = {0.f, 0.f, 0.f, 0.f, 0.f, 0.f, 0.f, 0.f};
    v8f acc[8];
#pragma unroll
    for (int i = 0; i < 8; ++i) acc[i] = zero;

    stageA(0, 0);
    stageB(0, 0);

    const int am = lane & 15;
    const int ah = lane >> 4;

    for (int k0 = 0; k0 < K; k0 += 32) {
        const int cur = (k0 >> 5) & 1;
        asm volatile("s_wait_asynccnt 0x0" ::: "memory");
        __syncthreads();                         // buf[cur] ready, prev reads done
        if (k0 + 32 < K) { stageA(cur ^ 1, k0 + 32); stageB(cur ^ 1, k0 + 32); }

        // A fragments: 4 x 16x32 rows of this wave's 64-row strip
        ushort16 av[4];
#pragma unroll
        for (int i = 0; i < 4; ++i) {
            const unsigned short* p = &ldsA[cur][(wm0 + i * 16 + am) * 32];
            ushort8 lo = *(const ushort8*)(p + ah * 8);
            ushort8 hi = *(const ushort8*)(p + 16 + ah * 8);
            av[i] = __builtin_shufflevector(lo, hi,
                      0, 1, 2, 3, 4, 5, 6, 7, 8, 9, 10, 11, 12, 13, 14, 15);
        }
        // B fragments: 2 x 32x16 columns of this wave's 32-col strip
        ushort16 bv[2];
#pragma unroll
        for (int j = 0; j < 2; ++j) {
            const unsigned short* q = &ldsB[cur][(wn0 + j * 16 + am) * 32];
            ushort8 blo = *(const ushort8*)(q + ah * 8);
            ushort8 bhi = *(const ushort8*)(q + 16 + ah * 8);
            bv[j] = __builtin_shufflevector(blo, bhi,
                      0, 1, 2, 3, 4, 5, 6, 7, 8, 9, 10, 11, 12, 13, 14, 15);
        }

#pragma unroll
        for (int i = 0; i < 4; ++i)
#pragma unroll
            for (int j = 0; j < 2; ++j)
                acc[i * 2 + j] = __builtin_amdgcn_wmma_f32_16x16x32_bf16(
                    false, __builtin_bit_cast(v16bf, av[i]),
                    false, __builtin_bit_cast(v16bf, bv[j]),
                    (short)0, acc[i * 2 + j], false, false);
    }

    // ---- epilogue: D tile layout VGPR r -> (M = r + 8*(lane>>4), N = lane&15)
    const int nl = lane & 15;
    const int mo = (lane >> 4) * 8;
#pragma unroll
    for (int i = 0; i < 4; ++i) {
#pragma unroll
        for (int j = 0; j < 2; ++j) {
            int gn = n0 + wn0 + j * 16 + nl;
            if (gn < N) {
                float bval = 0.f;
                if constexpr (HAS_BIAS) bval = bias[gn];
#pragma unroll
                for (int r = 0; r < 8; ++r) {
                    int gm = m0 + wm0 + i * 16 + mo + r;
                    float v = acc[i * 2 + j][r];
                    if constexpr (HAS_BIAS) v += bval;
                    if constexpr (EPI == 1)
                        v = (gn <= gm) ? v * scale : -3.0e38f;  // causal mask
                    C[(size_t)gm * ldc + gn] = v;
                }
            }
        }
    }
}

// ---------------------------------------------------------------------------
// f32 -> bf16 bulk conversion (4 elements / thread, n % 4 == 0)
// ---------------------------------------------------------------------------
__global__ __launch_bounds__(256) void cvt_f32_bf16_kernel(
    const float* __restrict__ in, unsigned short* __restrict__ out, int n4)
{
    int i = blockIdx.x * 256 + threadIdx.x;
    if (i < n4) {
        float4 f = ((const float4*)in)[i];
        uint2 p;
        p.x = (unsigned)f2bf(f.x) | ((unsigned)f2bf(f.y) << 16);
        p.y = (unsigned)f2bf(f.z) | ((unsigned)f2bf(f.w) << 16);
        ((uint2*)out)[i] = p;
    }
}

// ---------------------------------------------------------------------------
// RMSNorm row: out_bf16 = g * x * rsqrt(mean(x^2) + 1e-6)
// ---------------------------------------------------------------------------
__global__ __launch_bounds__(256) void rmsnorm_kernel(
    const float* __restrict__ in, int in_ld, int D,
    const float* __restrict__ g, unsigned short* __restrict__ out, int out_ld)
{
    __shared__ float red[256];
    const int t = threadIdx.x;
    const float* x = in + (size_t)blockIdx.x * in_ld;
    float s = 0.f;
    for (int i = t; i < D; i += 256) { float v = x[i]; s += v * v; }
    red[t] = s;
    __syncthreads();
    for (int o = 128; o > 0; o >>= 1) {
        if (t < o) red[t] += red[t + o];
        __syncthreads();
    }
    float scale = rsqrtf(red[0] / (float)D + 1e-6f);
    unsigned short* y = out + (size_t)blockIdx.x * out_ld;
    for (int i = t; i < D; i += 256) y[i] = f2bf(g[i] * x[i] * scale);
}

// ---------------------------------------------------------------------------
// RoPE on k_rope (cols 512..575 of kv_down_raw) -> f32 [S][64]
//   out[j]    = x[2j]*cos - x[2j+1]*sin
//   out[j+32] = x[2j+1]*cos + x[2j]*sin,  angle = pos * 10000^(-j/32)
// ---------------------------------------------------------------------------
__global__ void rope_k_kernel(const float* __restrict__ kv_raw,
                              const int* __restrict__ pos_ids,
                              float* __restrict__ krope)
{
    const int s = blockIdx.x;
    const int j = threadIdx.x;                    // one wave32
    float pos = (float)pos_ids[s];
    float ang = pos * expf(-(float)j * 0.28782313662425572f); // ln(1e4)/32
    float c = cosf(ang), sn = sinf(ang);
    float x0 = kv_raw[(size_t)s * 576 + 512 + 2 * j];
    float x1 = kv_raw[(size_t)s * 576 + 513 + 2 * j];
    krope[(size_t)s * 64 + j]      = x0 * c - x1 * sn;
    krope[(size_t)s * 64 + 32 + j] = x1 * c + x0 * sn;
}

// ---------------------------------------------------------------------------
// q_all[h][s][192] bf16: nope copy + RoPE on last 64 dims. Block per (h,s).
// ---------------------------------------------------------------------------
__global__ void build_q_kernel(const float* __restrict__ q_raw,
                               const int* __restrict__ pos_ids,
                               unsigned short* __restrict__ q_all)
{
    const int hs = blockIdx.x;
    const int h  = hs >> 11;
    const int s  = hs & (S_LEN - 1);
    const float* src = q_raw + (size_t)s * (HEADS * QD_D) + h * QD_D;
    unsigned short* dst = q_all + (size_t)hs * QD_D;
    const int t = threadIdx.x;
    if (t < NOPE_D) {
        dst[t] = f2bf(src[t]);
    } else if (t < NOPE_D + 32) {
        int j = t - NOPE_D;
        float pos = (float)pos_ids[s];
        float ang = pos * expf(-(float)j * 0.28782313662425572f);
        float c = cosf(ang), sn = sinf(ang);
        float x0 = src[NOPE_D + 2 * j];
        float x1 = src[NOPE_D + 2 * j + 1];
        dst[NOPE_D + j]      = f2bf(x0 * c - x1 * sn);
        dst[NOPE_D + 32 + j] = f2bf(x1 * c + x0 * sn);
    }
}

// ---------------------------------------------------------------------------
// k_all[h][s][192] = [k_nope | roped k_rope bcast]; vT[h][d][s] = V^T (bf16)
// ---------------------------------------------------------------------------
__global__ void build_kv_kernel(const float* __restrict__ kv_raw,
                                const float* __restrict__ krope,
                                unsigned short* __restrict__ k_all,
                                unsigned short* __restrict__ vT)
{
    const int hs = blockIdx.x;
    const int h  = hs >> 11;
    const int s  = hs & (S_LEN - 1);
    const float* src = kv_raw + (size_t)s * (HEADS * (NOPE_D + VD_D)) + h * (NOPE_D + VD_D);
    unsigned short* kd = k_all + (size_t)hs * QD_D;
    const int t = threadIdx.x;
    if (t < NOPE_D) {
        kd[t] = f2bf(src[t]);                                           // k_nope
        vT[((size_t)h * VD_D + t) * S_LEN + s] = f2bf(src[NOPE_D + t]); // V^T
    } else {
        int j = t - NOPE_D;                                             // 0..63
        kd[NOPE_D + j] = f2bf(krope[(size_t)s * ROPE_D + j]);
    }
}

// ---------------------------------------------------------------------------
// Row softmax in place over f32 rows of length 2048 (attn output in d_out)
// ---------------------------------------------------------------------------
__global__ __launch_bounds__(256) void softmax_kernel(float* __restrict__ attn)
{
    __shared__ float red[256];
    float* row = attn + (size_t)blockIdx.x * S_LEN;
    const int t = threadIdx.x;
    float mx = -3.4e38f;
    for (int i = t; i < S_LEN; i += 256) mx = fmaxf(mx, row[i]);
    red[t] = mx;
    __syncthreads();
    for (int o = 128; o > 0; o >>= 1) {
        if (t < o) red[t] = fmaxf(red[t], red[t + o]);
        __syncthreads();
    }
    mx = red[0];
    __syncthreads();
    float s = 0.f;
    for (int i = t; i < S_LEN; i += 256) {
        float e = expf(row[i] - mx);
        row[i] = e;
        s += e;
    }
    red[t] = s;
    __syncthreads();
    for (int o = 128; o > 0; o >>= 1) {
        if (t < o) red[t] += red[t + o];
        __syncthreads();
    }
    float inv = 1.0f / red[0];
    for (int i = t; i < S_LEN; i += 256) row[i] *= inv;
}

// ---------------------------------------------------------------------------
// Host orchestration.
// ---------------------------------------------------------------------------
extern "C" void kernel_launch(void* const* d_in, const int* in_sizes, int n_in,
                              void* d_out, int out_size, void* d_ws, size_t ws_size,
                              hipStream_t stream)
{
    (void)in_sizes; (void)n_in; (void)out_size; (void)ws_size;

    const float* hidden   = (const float*)d_in[0];
    const int*   pos      = (const int*)d_in[1];
    /* d_in[2] causal_mask recomputed analytically */
    const float* Wq_down  = (const float*)d_in[3];
    const float* bq_down  = (const float*)d_in[4];
    const float* gq       = (const float*)d_in[5];
    const float* Wq_up    = (const float*)d_in[6];
    const float* Wkv_down = (const float*)d_in[7];
    const float* bkv      = (const float*)d_in[8];
    const float* gkv      = (const float*)d_in[9];
    const float* Wkv_up   = (const float*)d_in[10];
    const float* Wo       = (const float*)d_in[11];

    float* out  = (float*)d_out;
    float* attn = out + (size_t)S_LEN * 2048;      // [16][2048][2048] f32

    // ---- workspace carve-up ----
    char* w = (char*)d_ws;
    float* R = (float*)w;  w += (size_t)S_LEN * 4096 * sizeof(float);       // 33.5 MB reused
    unsigned short* q_c   = (unsigned short*)w; w += (size_t)S_LEN * 1536 * 2;
    unsigned short* ckv   = (unsigned short*)w; w += (size_t)S_LEN * 512 * 2;
    float*          krope = (float*)w;          w += (size_t)S_LEN * ROPE_D * 4;
    unsigned short* q_all = (unsigned short*)w; w += (size_t)HEADS * S_LEN * QD_D * 2;
    unsigned short* k_all = (unsigned short*)w; w += (size_t)HEADS * S_LEN * QD_D * 2;
    unsigned short* vT    = (unsigned short*)w; w += (size_t)HEADS * VD_D * S_LEN * 2;
    unsigned short* Xb    = (unsigned short*)w; w += (size_t)S_LEN * 2048 * 2;
    unsigned short* Wqd_b = (unsigned short*)w; w += (size_t)1536 * 2048 * 2;
    unsigned short* Wkvd_b= (unsigned short*)w; w += (size_t)576 * 2048 * 2;
    unsigned short* Wqu_b = (unsigned short*)w; w += (size_t)3072 * 1536 * 2;
    unsigned short* Wkvu_b= (unsigned short*)w; w += (size_t)4096 * 512 * 2;
    unsigned short* Wo_b  = (unsigned short*)w;

    const dim3 blk(256);
    const long long HS_Q = (long long)S_LEN * QD_D;
    const long long HS_A = (long long)S_LEN * S_LEN;
    const long long HS_V = (long long)VD_D * S_LEN;

    // 0) one-shot bf16 conversions (activations + weights)
    auto cvt = [&](const float* src, unsigned short* dst, size_t n) {
        int n4 = (int)(n >> 2);
        cvt_f32_bf16_kernel<<<(n4 + 255) / 256, blk, 0, stream>>>(src, dst, n4);
    };
    cvt(hidden,   Xb,     (size_t)S_LEN * 2048);
    cvt(Wq_down,  Wqd_b,  (size_t)1536 * 2048);
    cvt(Wkv_down, Wkvd_b, (size_t)576 * 2048);
    cvt(Wq_up,    Wqu_b,  (size_t)3072 * 1536);
    cvt(Wkv_up,   Wkvu_b, (size_t)4096 * 512);
    cvt(Wo,       Wo_b,   (size_t)2048 * 2048);

    // 1) q_down_raw = X @ Wq_down^T + bq               [2048 x 1536]
    wmma_gemm<unsigned short, true, 0><<<dim3(12, 16, 1), blk, 0, stream>>>(
        Xb, 2048, 0, Wqd_b, 2048, 0, bq_down, R, 1536, 0, 1536, 2048, 1.0f);
    // 2) q_c = rmsnorm(q_down_raw, gq)
    rmsnorm_kernel<<<S_LEN, blk, 0, stream>>>(R, 1536, 1536, gq, q_c, 1536);

    // 3) kv_down_raw = X @ Wkv_down^T + bkv            [2048 x 576]
    wmma_gemm<unsigned short, true, 0><<<dim3(5, 16, 1), blk, 0, stream>>>(
        Xb, 2048, 0, Wkvd_b, 2048, 0, bkv, R, 576, 0, 576, 2048, 1.0f);
    // 4) ckv = rmsnorm(kv_down_raw[:, :512]); krope = rope(kv_down_raw[:, 512:])
    rmsnorm_kernel<<<S_LEN, blk, 0, stream>>>(R, 576, 512, gkv, ckv, 512);
    rope_k_kernel<<<S_LEN, 32, 0, stream>>>(R, pos, krope);

    // 5) q_raw = q_c @ Wq_up^T                         [2048 x 3072]
    wmma_gemm<unsigned short, false, 0><<<dim3(24, 16, 1), blk, 0, stream>>>(
        q_c, 1536, 0, Wqu_b, 1536, 0, nullptr, R, 3072, 0, 3072, 1536, 1.0f);
    // 6) q_all = [q_nope | rope(q_rope)]
    build_q_kernel<<<HEADS * S_LEN, 192, 0, stream>>>(R, pos, q_all);

    // 7) kv_raw = ckv @ Wkv_up^T                       [2048 x 4096]
    wmma_gemm<unsigned short, false, 0><<<dim3(32, 16, 1), blk, 0, stream>>>(
        ckv, 512, 0, Wkvu_b, 512, 0, nullptr, R, 4096, 0, 4096, 512, 1.0f);
    // 8) k_all, vT
    build_kv_kernel<<<HEADS * S_LEN, 192, 0, stream>>>(R, krope, k_all, vT);

    // 9) scores = (q_all @ k_all^T)/sqrt(192), causal mask -> attn (f32, d_out)
    wmma_gemm<unsigned short, false, 1><<<dim3(16, 16, HEADS), blk, 0, stream>>>(
        q_all, QD_D, HS_Q, k_all, QD_D, HS_Q, nullptr, attn, S_LEN, HS_A,
        S_LEN, QD_D, 0.07216878364870323f);
    // 10) softmax rows in place
    softmax_kernel<<<HEADS * S_LEN, blk, 0, stream>>>(attn);

    // 11) ctx[s][h*128+d] = attn[h] @ v[h]   (A = f32 attn, B = bf16 V^T)
    wmma_gemm<float, false, 0><<<dim3(1, 16, HEADS), blk, 0, stream>>>(
        attn, S_LEN, HS_A, vT, S_LEN, HS_V, nullptr, R, 2048, VD_D,
        VD_D, S_LEN, 1.0f);

    // 12) out = ctx @ Wo^T                             [2048 x 2048]
    wmma_gemm<float, false, 0><<<dim3(16, 16, 1), blk, 0, stream>>>(
        R, 2048, 0, Wo_b, 2048, 0, nullptr, out, 2048, 0, 2048, 2048, 1.0f);
}
